// ProbSparseAttention_63316407878076
// MI455X (gfx1250) — matrix operand order repaired and checked by reading, not compiled, not run on hardware
//
#include <hip/hip_runtime.h>
#include <hip/hip_bf16.h>
#include <stdint.h>

// ---------------------------------------------------------------------------
// Problem constants (from reference): B=4, L=4096, D=1024, H=16, D_K=64, U=41
// ---------------------------------------------------------------------------
#define BB   4
#define LL   4096
#define DD   1024
#define HH   16
#define DK   64
#define UU   41
#define UP   48          // U padded to 3 WMMA tiles of 16
#define BH   (BB*HH)     // 64
#define MTOT (BB*LL)     // 16384 rows in the projection GEMMs

typedef __bf16 bf16_t;
typedef __attribute__((ext_vector_type(16))) __bf16 v16bf;
typedef __attribute__((ext_vector_type(8)))  float  v8f;

__device__ __forceinline__ v8f zero8() {
    v8f z = {0.f,0.f,0.f,0.f,0.f,0.f,0.f,0.f};
    return z;
}

__device__ __forceinline__ v8f wmma_bf16(v16bf a, v16bf b, v8f c) {
    // D = A(16x32 bf16) * B(32x16 bf16) + C(16x16 f32)
    return __builtin_amdgcn_wmma_f32_16x16x32_bf16(
        false, a, false, b, (short)0, c, false, false);
}

union FragU { uint4 q[2]; v16bf v; };

// A fragment, 16x32 bf16, source row-major (row, k) with leading dim `ld` (elems).
// ISA 7.12.2: lanes 0-15 -> M=lane, v0..3 = K0..7, v4..7 = K16..23
//             lanes 16-31 -> M=lane-16, v0..3 = K8..15, v4..7 = K24..31
__device__ __forceinline__ v16bf load_a_frag(const bf16_t* base, int ld) {
    int lane = threadIdx.x & 31;
    int row  = lane & 15;
    int kh   = lane >> 4;                 // 0 or 1
    const bf16_t* p = base + (size_t)row * ld + kh * 8;
    FragU f;
    f.q[0] = *(const uint4*)(p);          // K +0..7
    f.q[1] = *(const uint4*)(p + 16);     // K +16..23
    return f.v;
}

// B fragment, 32x16 bf16 (KxN), source stored N-major: element (n,k) at base[n*ld+k].
// lanes 0-15 -> N=lane, v0..3 = K0..7, v4..7 = K16..23 ; lanes 16-31 -> N=lane-16, v0..3 = K8..15, v4..7 = K24..31
__device__ __forceinline__ v16bf load_b_frag(const bf16_t* base, int ld) {
    int lane = threadIdx.x & 31;
    int n    = lane & 15;
    int kh   = lane >> 4;
    const bf16_t* p = base + (size_t)n * ld + kh * 8;
    FragU f;
    f.q[0] = *(const uint4*)(p);          // K +0..7
    f.q[1] = *(const uint4*)(p + 16);     // K +16..23
    return f.v;
}

// Async 16-byte global -> LDS copy (CDNA5 ASYNCcnt path, GVS addressing).
__device__ __forceinline__ void async_cp16(uint32_t lds_off, const void* base,
                                           uint32_t byte_off) {
    uint64_t b = (uint64_t)(uintptr_t)base;
    asm volatile("global_load_async_to_lds_b128 %0, %1, %2"
                 :: "v"(lds_off), "v"(byte_off), "s"(b) : "memory");
}
__device__ __forceinline__ void wait_async0() {
    asm volatile("s_wait_asynccnt 0" ::: "memory");
}

// ---------------------------------------------------------------------------
// fp32 -> bf16 conversion kernels
// ---------------------------------------------------------------------------
__global__ void cvt_bf16_kernel(const float* __restrict__ src,
                                bf16_t* __restrict__ dst, int n) {
    int i = blockIdx.x * blockDim.x + threadIdx.x;
    if (i < n) dst[i] = (bf16_t)src[i];
}

// W (K x N, row-major) -> Wt (N x K) bf16, so B-fragments read contiguous K.
__global__ void transpose_bf16_kernel(const float* __restrict__ src,
                                      bf16_t* __restrict__ dst) {
    int i = blockIdx.x * blockDim.x + threadIdx.x;  // over DD*DD
    int k = i >> 10, n = i & (DD - 1);
    dst[(size_t)n * DD + k] = (bf16_t)src[i];
}

// ---------------------------------------------------------------------------
// Big bf16 WMMA GEMM: out = A[M x 1024] * W[1024 x 1024] + bias
//   A row-major bf16, Bt stored N-major bf16 (Bt[n*1024+k]).
//   mode 0: fp32 row-major [M x 1024]   (final output)
//   mode 1: bf16 head layout [B,H,L,DK] (Q, K)
//   mode 2: bf16 transposed head layout [B,H,DK,L] (V -> Vt)
// Block 256 thr (8 waves), block tile 128x64, wave tile 32x32, k-step 32.
// Tiles staged in LDS with global_load_async_to_lds_b128.
// ---------------------------------------------------------------------------
__global__ __launch_bounds__(256) void gemm_bf16_kernel(
    const bf16_t* __restrict__ A, const bf16_t* __restrict__ Bt,
    const float* __restrict__ bias, void* __restrict__ outp, int mode) {
    __shared__ __align__(16) bf16_t sA[128 * 32];   // [row][k]
    __shared__ __align__(16) bf16_t sB[64 * 32];    // [n][k]
    int tid  = threadIdx.x;
    int lane = tid & 31, wave = tid >> 5;
    int m0 = blockIdx.x * 128;
    int n0 = blockIdx.y * 64;
    int wm = (wave & 3) * 32;          // 4 waves along M
    int wn = (wave >> 2) * 32;         // 2 waves along N

    v8f acc[2][2];
    for (int i = 0; i < 2; ++i)
        for (int j = 0; j < 2; ++j) acc[i][j] = zero8();

    uint32_t ldsA = (uint32_t)(uintptr_t)(void*)sA;
    uint32_t ldsB = (uint32_t)(uintptr_t)(void*)sB;

    for (int k0 = 0; k0 < DD; k0 += 32) {
        __syncthreads();   // previous iteration's LDS reads are done
        // A tile: 128 rows x 32 k = 512 x 16B chunks
#pragma unroll
        for (int i = 0; i < 2; ++i) {
            int idx = tid + i * 256;
            int r = idx >> 2, c = idx & 3;
            async_cp16(ldsA + (uint32_t)(r * 64 + c * 16), A,
                       (uint32_t)(((m0 + r) * DD + k0 + c * 8) * 2));
        }
        // B tile: 64 n-rows x 32 k = 256 x 16B chunks
        {
            int r = tid >> 2, c = tid & 3;
            async_cp16(ldsB + (uint32_t)(r * 64 + c * 16), Bt,
                       (uint32_t)(((n0 + r) * DD + k0 + c * 8) * 2));
        }
        wait_async0();
        __syncthreads();

        v16bf bf0 = load_b_frag(sB + (wn + 0) * 32, 32);
        v16bf bf1 = load_b_frag(sB + (wn + 16) * 32, 32);
#pragma unroll
        for (int mi = 0; mi < 2; ++mi) {
            v16bf af = load_a_frag(sA + (wm + mi * 16) * 32, 32);
            acc[mi][0] = wmma_bf16(af, bf0, acc[mi][0]);
            acc[mi][1] = wmma_bf16(af, bf1, acc[mi][1]);
        }
    }

    // Epilogue: C layout = VGPR r -> M = r (lanes 0-15) / 8+r (lanes 16-31), N = lane&15
    int cn = lane & 15;
    int rb = (lane >> 4) * 8;
#pragma unroll
    for (int mi = 0; mi < 2; ++mi) {
#pragma unroll
        for (int ni = 0; ni < 2; ++ni) {
            int gn = n0 + wn + ni * 16 + cn;
            float bv = bias[gn];
#pragma unroll
            for (int r = 0; r < 8; ++r) {
                int gm = m0 + wm + mi * 16 + rb + r;
                float val = acc[mi][ni][r] + bv;
                if (mode == 0) {
                    ((float*)outp)[(size_t)gm * DD + gn] = val;
                } else {
                    int b = gm >> 12, l = gm & (LL - 1);
                    int h = gn >> 6,  dk = gn & (DK - 1);
                    if (mode == 1)
                        ((bf16_t*)outp)[(((size_t)(b * HH + h)) * LL + l) * DK + dk] =
                            (bf16_t)val;
                    else
                        ((bf16_t*)outp)[(((size_t)(b * HH + h)) * DK + dk) * LL + l] =
                            (bf16_t)val;
                }
            }
        }
    }
}

// ---------------------------------------------------------------------------
// Score kernel: Mscore[bh][l] = max_u(QKs) - mean_u(QKs), QKs = Q . K_sample^T
// Block 256 (8 waves), each wave does 16 l-rows via WMMA against 48-padded Ks.
// ---------------------------------------------------------------------------
__global__ __launch_bounds__(256) void score_kernel(
    const bf16_t* __restrict__ Qh, const bf16_t* __restrict__ Kh,
    const int* __restrict__ kidx, float* __restrict__ Mscore) {
    int bh = blockIdx.x;
    int l0 = blockIdx.y * 128;
    __shared__ __align__(16) bf16_t sKs[UP * DK];       // [u][dk], zero-padded
    __shared__ float sQK[8][16 * UP];
    int tid = threadIdx.x;

    for (int idx = tid; idx < UP * DK / 8; idx += 256) {  // 16B chunks
        int u = idx >> 3, c = idx & 7;
        uint4 v = {0u, 0u, 0u, 0u};
        if (u < UU) {
            int l = kidx[u];
            v = *(const uint4*)(Kh + ((size_t)bh * LL + l) * DK + c * 8);
        }
        *(uint4*)(sKs + u * DK + c * 8) = v;
    }
    __syncthreads();

    int wave = tid >> 5, lane = tid & 31;
    int lrow = l0 + wave * 16;
    const bf16_t* Abase = Qh + ((size_t)bh * LL + lrow) * DK;
    v16bf a0 = load_a_frag(Abase, DK);
    v16bf a1 = load_a_frag(Abase + 32, DK);

    int cn = lane & 15, rb = (lane >> 4) * 8;
#pragma unroll
    for (int nt = 0; nt < 3; ++nt) {
        v16bf b0 = load_b_frag(sKs + nt * 16 * DK, DK);
        v16bf b1 = load_b_frag(sKs + nt * 16 * DK + 32, DK);
        v8f acc = zero8();
        acc = wmma_bf16(a0, b0, acc);
        acc = wmma_bf16(a1, b1, acc);
#pragma unroll
        for (int r = 0; r < 8; ++r)
            sQK[wave][(rb + r) * UP + nt * 16 + cn] = acc[r];
    }
    __syncthreads();

    if (lane < 16) {
        float mx = -3.4e38f, sum = 0.f;
        for (int u = 0; u < UU; ++u) {
            float v = sQK[wave][lane * UP + u];
            mx = fmaxf(mx, v);
            sum += v;
        }
        Mscore[(size_t)bh * LL + lrow + lane] = mx - sum * (1.0f / UU);
    }
}

// ---------------------------------------------------------------------------
// Top-41 per (b,h): iterative block-wide argmax over 4096 scores.
// ---------------------------------------------------------------------------
__global__ __launch_bounds__(256) void topk_kernel(
    const float* __restrict__ Mscore, int* __restrict__ top_idx) {
    int bh = blockIdx.x;
    __shared__ float vals[LL];
    __shared__ float rmax[256];
    __shared__ int   ridx[256];
    int tid = threadIdx.x;
    for (int i = tid; i < LL; i += 256) vals[i] = Mscore[(size_t)bh * LL + i];
    __syncthreads();
    for (int it = 0; it < UU; ++it) {
        float bm = -3.4e38f; int bi = 0;
        for (int i = tid; i < LL; i += 256) {
            float v = vals[i];
            if (v > bm) { bm = v; bi = i; }
        }
        rmax[tid] = bm; ridx[tid] = bi;
        __syncthreads();
        for (int s = 128; s > 0; s >>= 1) {
            if (tid < s && rmax[tid + s] > rmax[tid]) {
                rmax[tid] = rmax[tid + s]; ridx[tid] = ridx[tid + s];
            }
            __syncthreads();
        }
        if (tid == 0) {
            top_idx[bh * UU + it] = ridx[0];
            vals[ridx[0]] = -3.4e38f;
        }
        __syncthreads();
    }
}

// ---------------------------------------------------------------------------
// Sparse scores: S[bh][u(48)][l] = (Q_sparse . K^T) / 8. Block 128 (4 waves),
// each wave covers 64 columns of L; Q_sparse gathered into LDS once.
// ---------------------------------------------------------------------------
__global__ __launch_bounds__(128) void sparse_scores_kernel(
    const bf16_t* __restrict__ Qh, const bf16_t* __restrict__ Kh,
    const int* __restrict__ top_idx, float* __restrict__ S) {
    int bh = blockIdx.x;
    int l0 = blockIdx.y * 256;
    __shared__ __align__(16) bf16_t sQ[UP * DK];
    int tid = threadIdx.x;
    for (int idx = tid; idx < UP * DK / 8; idx += 128) {
        int u = idx >> 3, c = idx & 7;
        uint4 v = {0u, 0u, 0u, 0u};
        if (u < UU) {
            int l = top_idx[bh * UU + u];
            v = *(const uint4*)(Qh + ((size_t)bh * LL + l) * DK + c * 8);
        }
        *(uint4*)(sQ + u * DK + c * 8) = v;
    }
    __syncthreads();

    int wave = tid >> 5, lane = tid & 31;
    v16bf a[3][2];
#pragma unroll
    for (int mt = 0; mt < 3; ++mt) {
        a[mt][0] = load_a_frag(sQ + mt * 16 * DK, DK);
        a[mt][1] = load_a_frag(sQ + mt * 16 * DK + 32, DK);
    }
    int nbase = l0 + wave * 64;
    int cn = lane & 15, rb = (lane >> 4) * 8;
#pragma unroll
    for (int nt = 0; nt < 4; ++nt) {
        const bf16_t* Bb = Kh + ((size_t)bh * LL + nbase + nt * 16) * DK;
        v16bf b0 = load_b_frag(Bb, DK);
        v16bf b1 = load_b_frag(Bb + 32, DK);
#pragma unroll
        for (int mt = 0; mt < 3; ++mt) {
            v8f acc = zero8();
            acc = wmma_bf16(a[mt][0], b0, acc);
            acc = wmma_bf16(a[mt][1], b1, acc);
            int col = nbase + nt * 16 + cn;
#pragma unroll
            for (int r = 0; r < 8; ++r) {
                int u = mt * 16 + rb + r;
                S[((size_t)bh * UP + u) * LL + col] = acc[r] * 0.125f;
            }
        }
    }
}

// ---------------------------------------------------------------------------
// Row softmax over L=4096 (one wave32 per row), bf16 output.
// ---------------------------------------------------------------------------
__global__ __launch_bounds__(256) void softmax_kernel(
    const float* __restrict__ S, bf16_t* __restrict__ attnb) {
    int row  = blockIdx.x * 8 + (threadIdx.x >> 5);   // over BH*48
    int lane = threadIdx.x & 31;
    const float* src = S + (size_t)row * LL;
    float mx = -3.4e38f;
    for (int i = lane; i < LL; i += 32) mx = fmaxf(mx, src[i]);
    for (int o = 16; o > 0; o >>= 1) mx = fmaxf(mx, __shfl_xor(mx, o, 32));
    float sum = 0.f;
    for (int i = lane; i < LL; i += 32) sum += __expf(src[i] - mx);
    for (int o = 16; o > 0; o >>= 1) sum += __shfl_xor(sum, o, 32);
    float inv = 1.0f / sum;
    bf16_t* dst = attnb + (size_t)row * LL;
    for (int i = lane; i < LL; i += 32)
        dst[i] = (bf16_t)(__expf(src[i] - mx) * inv);
}

// ---------------------------------------------------------------------------
// sparse_out[bh][u][dk] = attn @ V  (contraction over L via WMMA; V stored
// transposed [bh][dk][l] so B-fragments are contiguous).
// Block 128 (4 waves), wave w owns dk-tile w; 3 u-tiles, 128 k-steps of 32.
// ---------------------------------------------------------------------------
__global__ __launch_bounds__(128) void attnv_kernel(
    const bf16_t* __restrict__ attnb, const bf16_t* __restrict__ Vt,
    float* __restrict__ sparse_out) {
    int bh = blockIdx.x;
    int wave = threadIdx.x >> 5, lane = threadIdx.x & 31;
    v8f acc[3] = {zero8(), zero8(), zero8()};
    const bf16_t* Arow = attnb + (size_t)bh * UP * LL;
    const bf16_t* Bb   = Vt + ((size_t)bh * DK + wave * 16) * LL;
    for (int k0 = 0; k0 < LL; k0 += 32) {
        v16bf b0 = load_b_frag(Bb + k0, LL);
#pragma unroll
        for (int mt = 0; mt < 3; ++mt) {
            v16bf af = load_a_frag(Arow + (size_t)mt * 16 * LL + k0, LL);
            acc[mt] = wmma_bf16(af, b0, acc[mt]);
        }
    }
    int cn = lane & 15, rb = (lane >> 4) * 8;
#pragma unroll
    for (int mt = 0; mt < 3; ++mt)
#pragma unroll
        for (int r = 0; r < 8; ++r) {
            int u = mt * 16 + rb + r;
            sparse_out[((size_t)bh * UP + u) * DK + wave * 16 + cn] = acc[mt][r];
        }
}

// ---------------------------------------------------------------------------
// meanV[bh][dk] = mean over L of V (one wave per (bh,dk) row of Vt).
// ---------------------------------------------------------------------------
__global__ __launch_bounds__(256) void meanv_kernel(
    const bf16_t* __restrict__ Vt, float* __restrict__ meanV) {
    int row  = blockIdx.x * 8 + (threadIdx.x >> 5);   // over BH*DK = 4096
    int lane = threadIdx.x & 31;
    const bf16_t* src = Vt + (size_t)row * LL;
    float sum = 0.f;
    for (int i = lane; i < LL; i += 32) sum += (float)src[i];
    for (int o = 16; o > 0; o >>= 1) sum += __shfl_xor(sum, o, 32);
    if (lane == 0) meanV[row] = sum * (1.0f / LL);
}

// ctx[b,l,(h,dk)] = meanV[bh][dk] broadcast, bf16
__global__ void ctx_fill_kernel(const float* __restrict__ meanV,
                                bf16_t* __restrict__ ctx) {
    int i = blockIdx.x * blockDim.x + threadIdx.x;    // over MTOT*DD
    int col = i & (DD - 1);
    int b   = i >> 22;                                 // / (LL*DD)
    ctx[i] = (bf16_t)meanV[((b * HH) + (col >> 6)) * DK + (col & (DK - 1))];
}

// Scatter sparse_out rows into ctx at top_idx positions.
__global__ __launch_bounds__(256) void scatter_kernel(
    const float* __restrict__ sparse_out, const int* __restrict__ top_idx,
    bf16_t* __restrict__ ctx) {
    int bh = blockIdx.x;
    int b = bh >> 4, h = bh & (HH - 1);
    for (int i = threadIdx.x; i < UU * DK; i += 256) {
        int u = i >> 6, dk = i & (DK - 1);
        int l = top_idx[bh * UU + u];
        ctx[((size_t)(b * LL + l)) * DD + h * DK + dk] =
            (bf16_t)sparse_out[((size_t)bh * UP + u) * DK + dk];
    }
}

// ---------------------------------------------------------------------------
// Host-side orchestration
// ---------------------------------------------------------------------------
extern "C" void kernel_launch(void* const* d_in, const int* in_sizes, int n_in,
                              void* d_out, int out_size, void* d_ws, size_t ws_size,
                              hipStream_t stream) {
    const float* x    = (const float*)d_in[0];
    const float* W_q  = (const float*)d_in[1];
    const float* b_q  = (const float*)d_in[2];
    const float* W_k  = (const float*)d_in[3];
    const float* b_k  = (const float*)d_in[4];
    const float* W_v  = (const float*)d_in[5];
    const float* b_v  = (const float*)d_in[6];
    const float* W_o  = (const float*)d_in[7];
    const float* b_o  = (const float*)d_in[8];
    const int*   kidx = (const int*)d_in[9];

    char* ws = (char*)d_ws;
    size_t off = 0;
    auto alloc = [&](size_t bytes) -> char* {
        char* p = ws + off;
        off += (bytes + 255) & ~(size_t)255;
        return p;
    };
    bf16_t* xb    = (bf16_t*)alloc((size_t)MTOT * DD * 2);
    bf16_t* Wq_t  = (bf16_t*)alloc((size_t)DD * DD * 2);
    bf16_t* Wk_t  = (bf16_t*)alloc((size_t)DD * DD * 2);
    bf16_t* Wv_t  = (bf16_t*)alloc((size_t)DD * DD * 2);
    bf16_t* Wo_t  = (bf16_t*)alloc((size_t)DD * DD * 2);
    bf16_t* Qh    = (bf16_t*)alloc((size_t)BH * LL * DK * 2);
    bf16_t* Kh    = (bf16_t*)alloc((size_t)BH * LL * DK * 2);
    bf16_t* Vt    = (bf16_t*)alloc((size_t)BH * DK * LL * 2);
    float*  Msc   = (float*)alloc((size_t)BH * LL * 4);
    int*    tidx  = (int*)alloc((size_t)BH * UU * 4);
    float*  S     = (float*)alloc((size_t)BH * UP * LL * 4);
    bf16_t* attnb = (bf16_t*)alloc((size_t)BH * UP * LL * 2);
    float*  spout = (float*)alloc((size_t)BH * UP * DK * 4);
    float*  meanV = (float*)alloc((size_t)BH * DK * 4);
    bf16_t* ctx   = (bf16_t*)alloc((size_t)MTOT * DD * 2);

    // 1) bf16 conversions
    cvt_bf16_kernel<<<(MTOT * DD) / 256, 256, 0, stream>>>(x, xb, MTOT * DD);
    transpose_bf16_kernel<<<(DD * DD) / 256, 256, 0, stream>>>(W_q, Wq_t);
    transpose_bf16_kernel<<<(DD * DD) / 256, 256, 0, stream>>>(W_k, Wk_t);
    transpose_bf16_kernel<<<(DD * DD) / 256, 256, 0, stream>>>(W_v, Wv_t);
    transpose_bf16_kernel<<<(DD * DD) / 256, 256, 0, stream>>>(W_o, Wo_t);

    // 2) Q/K/V projection GEMMs (WMMA, async-LDS staged)
    dim3 ggrid(MTOT / 128, DD / 64);
    gemm_bf16_kernel<<<ggrid, 256, 0, stream>>>(xb, Wq_t, b_q, (void*)Qh, 1);
    gemm_bf16_kernel<<<ggrid, 256, 0, stream>>>(xb, Wk_t, b_k, (void*)Kh, 1);
    gemm_bf16_kernel<<<ggrid, 256, 0, stream>>>(xb, Wv_t, b_v, (void*)Vt, 2);

    // 3) ProbSparse score + top-41
    score_kernel<<<dim3(BH, LL / 128), 256, 0, stream>>>(Qh, Kh, kidx, Msc);
    topk_kernel<<<BH, 256, 0, stream>>>(Msc, tidx);

    // 4) sparse attention
    sparse_scores_kernel<<<dim3(BH, LL / 256), 128, 0, stream>>>(Qh, Kh, tidx, S);
    softmax_kernel<<<(BH * UP) / 8, 256, 0, stream>>>(S, attnb);
    attnv_kernel<<<BH, 128, 0, stream>>>(attnb, Vt, spout);

    // 5) context assembly
    meanv_kernel<<<(BH * DK) / 8, 256, 0, stream>>>(Vt, meanV);
    ctx_fill_kernel<<<(MTOT * DD) / 256, 256, 0, stream>>>(meanV, ctx);
    scatter_kernel<<<BH, 256, 0, stream>>>(spout, tidx, ctx);

    // 6) output projection into d_out (fp32)
    gemm_bf16_kernel<<<ggrid, 256, 0, stream>>>(ctx, Wo_t, b_o, d_out, 0);
    (void)in_sizes; (void)n_in; (void)out_size; (void)ws_size;
}
